// nBRC_40278203302324
// MI455X (gfx1250) — matrix operand order repaired and checked by reading, not compiled, not run on hardware
//
#include <hip/hip_runtime.h>

// ---------------------------------------------------------------------------
// nBRC recurrent cell for MI455X (gfx1250), wave32 + WMMA.
//   B=64, L=512, IN=512, MEM=1024
// Plan:
//   k_cvt   : f32 -> f16 copies of u, W_im, W_mm
//   k_init  : zero h double-buffer + grid-barrier counters
//   k_proj  : in_emb = u @ W_im^T   (M=32768,N=3072,K=512)  v_wmma_f32_16x16x32_f16
//   k_rec   : persistent 128-WG kernel, 512 sequential steps, W_mm slice in LDS,
//             one atomic grid barrier per step, h state in registers (f32).
// ---------------------------------------------------------------------------

typedef __attribute__((ext_vector_type(16))) _Float16 v16h;
typedef __attribute__((ext_vector_type(8)))  _Float16 v8h;
typedef __attribute__((ext_vector_type(8)))  float    v8f;

#define WMMA_F16F32(a, b, c) \
  __builtin_amdgcn_wmma_f32_16x16x32_f16(false, (a), false, (b), (short)0, (c), false, false)

static constexpr int  Bb   = 64;
static constexpr int  Ll   = 512;
static constexpr int  INf  = 512;
static constexpr int  MEM  = 1024;
static constexpr int  NWG  = 128;              // persistent workgroups (must be co-resident)

static constexpr size_t U_ELEMS   = (size_t)Bb * Ll * INf;      // 16,777,216
static constexpr size_t WIM_ELEMS = (size_t)3 * MEM * INf;      //  1,572,864
static constexpr size_t WMM_ELEMS = (size_t)2 * MEM * MEM;      //  2,097,152
static constexpr size_t EMB_ELEMS = (size_t)Bb * Ll * 3 * MEM;  // 100,663,296
static constexpr size_t H_ELEMS   = (size_t)2 * Bb * MEM;       // double-buffered h (f16)
static constexpr size_t SEQ_ELEMS = (size_t)Bb * Ll * MEM;      // 33,554,432 (output head)

// ---- fragment builders (contiguous-k layouts, 16-bit data, wave32) ------------
static __device__ __forceinline__ v16h frag_cat(v8h lo, v8h hi) {
  return __builtin_shufflevector(lo, hi, 0,1,2,3,4,5,6,7,8,9,10,11,12,13,14,15);
}

// A matrix 16x32 (M x K): lane = 16*half + m ; elem e -> K = 16*half + e
static __device__ __forceinline__ v16h load_frag_a(const _Float16* base, int ld,
                                                   int row0, int k0, int lane) {
  const int h = (lane >> 4) & 1, r = lane & 15;
  const _Float16* p = base + (size_t)(row0 + r) * ld + k0 + 16 * h;
  v8h lo = *(const v8h*)p;
  v8h hi = *(const v8h*)(p + 8);
  return frag_cat(lo, hi);
}

// B matrix 32x16 (K x N), B[k][n] = W[row0+n][k] (W row-major, K contiguous):
// lane = 16*half + n ; elem e -> K = k0 + 16*half + e
static __device__ __forceinline__ v16h load_frag_b(const _Float16* base, int ld,
                                                   int row0, int k0, int lane) {
  const int h = (lane >> 4) & 1, r = lane & 15;
  const _Float16* p = base + (size_t)(row0 + r) * ld + k0 + 16 * h;
  v8h lo = *(const v8h*)p;
  v8h hi = *(const v8h*)(p + 8);
  return frag_cat(lo, hi);
}

// ---- grid-wide barrier (persistent kernel; bar[0]=count, bar[1]=generation) -
static __device__ __forceinline__ void grid_barrier(unsigned* bar) {
  __threadfence();
  __syncthreads();
  if (threadIdx.x == 0) {
    unsigned g = __hip_atomic_load(&bar[1], __ATOMIC_RELAXED, __HIP_MEMORY_SCOPE_AGENT);
    unsigned a = __hip_atomic_fetch_add(&bar[0], 1u, __ATOMIC_ACQ_REL, __HIP_MEMORY_SCOPE_AGENT);
    if (a == (unsigned)NWG - 1u) {
      __hip_atomic_store(&bar[0], 0u, __ATOMIC_RELAXED, __HIP_MEMORY_SCOPE_AGENT);
      __hip_atomic_fetch_add(&bar[1], 1u, __ATOMIC_RELEASE, __HIP_MEMORY_SCOPE_AGENT);
    } else {
      while (__hip_atomic_load(&bar[1], __ATOMIC_ACQUIRE, __HIP_MEMORY_SCOPE_AGENT) == g)
        __builtin_amdgcn_s_sleep(2);
    }
  }
  __syncthreads();
  __threadfence();
}

// ---- f32 -> f16 conversion --------------------------------------------------
__global__ void k_cvt(const float* __restrict__ src, _Float16* __restrict__ dst, size_t n) {
  size_t i = (size_t)blockIdx.x * blockDim.x + threadIdx.x;
  size_t stride = (size_t)gridDim.x * blockDim.x;
  for (; i < n; i += stride) dst[i] = (_Float16)src[i];
}

// ---- init: zero h double-buffer + barrier -----------------------------------
__global__ void k_init(_Float16* __restrict__ hbuf, unsigned* __restrict__ bar) {
  size_t i = (size_t)blockIdx.x * blockDim.x + threadIdx.x;
  if (i < H_ELEMS) hbuf[i] = (_Float16)0.0f;
  if (i == 0) { bar[0] = 0u; bar[1] = 0u; }
}

// ---- input projection: emb[i][j] = sum_k u16[i][k] * wim16[j][k] ------------
// grid: (M/64, N/32), 256 threads = 8 waves, one 16x16 tile per wave.
__global__ __launch_bounds__(256) void k_proj(const _Float16* __restrict__ u16,
                                              const _Float16* __restrict__ wim16,
                                              _Float16* __restrict__ emb) {
  const int lane = threadIdx.x & 31;
  const int wave = threadIdx.x >> 5;
  const int h    = (lane >> 4) & 1, lr = lane & 15;
  const int m0 = blockIdx.x * 64 + (wave & 3) * 16;
  const int n0 = blockIdx.y * 32 + (wave >> 2) * 16;

  v8f acc = {};
#pragma unroll 4
  for (int k0 = 0; k0 < INf; k0 += 32) {
    v16h a = load_frag_a(u16,   INf, m0, k0, lane);
    v16h b = load_frag_b(wim16, INf, n0, k0, lane);
    acc = WMMA_F16F32(a, b, acc);
  }
  // C/D layout: elem v -> row m0 + v + 8*half, col n0 + lr
#pragma unroll
  for (int v = 0; v < 8; ++v)
    emb[(size_t)(m0 + v + 8 * h) * (3 * MEM) + n0 + lr] = (_Float16)acc[v];
}

// ---- persistent recurrent kernel -------------------------------------------
// WG wg owns h-columns [8wg, 8wg+8). Its 16-wide N-tile = 8 m_a cols + 8 m_c cols
// (W_mm rows 8wg.. and 1024+8wg..). W_mm slice cached in LDS for all 512 steps.
static constexpr int LDS_STRIDE = MEM + 8;     // 1032 halfs = 2064 B (conflict-free b128)

__global__ __launch_bounds__(128) void k_rec(const _Float16* __restrict__ wmm16,
                                             const _Float16* __restrict__ emb,
                                             _Float16* __restrict__ hbuf,   // 2 * 64*1024 f16
                                             float* __restrict__ out,
                                             unsigned* __restrict__ bar) {
  __shared__ __align__(16) _Float16 ldsB[16 * LDS_STRIDE];   // ~33 KB
  __shared__ float stage[64 * 17];                            // m tile staging, padded

  const int wg   = blockIdx.x;
  const int tid  = threadIdx.x;
  const int wave = tid >> 5;                 // 0..3 -> batch-row tile
  const int lane = tid & 31;
  const int h    = (lane >> 4) & 1, lr = lane & 15;

  // Load this WG's 16 x 1024 W_mm slice into LDS (rows 0-7: m_a, rows 8-15: m_c).
  for (int idx = tid; idx < 16 * 128; idx += 128) {          // 128 x 16B chunks per row
    const int r = idx >> 7, c = idx & 127;
    const int gr = (r < 8) ? (wg * 8 + r) : (MEM + wg * 8 + (r - 8));
    *(v8h*)&ldsB[r * LDS_STRIDE + c * 8] = *(const v8h*)&wmm16[(size_t)gr * MEM + c * 8];
  }

  float hreg[4] = {0.f, 0.f, 0.f, 0.f};      // this thread's 4 h elements (f32 master copy)
  __syncthreads();

  int p = 0;                                  // h read buffer
  for (int t = 0; t < Ll; ++t) {
    // ---- GEMM: tile rows [wave*16, +16), 16 mixed columns, K = 1024 ----
    const _Float16* hread = hbuf + (size_t)p * (Bb * MEM);
    v8f acc = {};
#pragma unroll 4
    for (int k0 = 0; k0 < MEM; k0 += 32) {
      v16h a = load_frag_a(hread, MEM, wave * 16, k0, lane);
      const _Float16* pb = &ldsB[lr * LDS_STRIDE + k0 + 16 * h];
      v16h b = frag_cat(*(const v8h*)pb, *(const v8h*)(pb + 8));
      acc = WMMA_F16F32(a, b, acc);
    }
#pragma unroll
    for (int v = 0; v < 8; ++v)
      stage[(wave * 16 + v + 8 * h) * 17 + lr] = acc[v];
    __syncthreads();

    // ---- elementwise nBRC update (all f32; h in registers) ----
    _Float16* hwrite = hbuf + (size_t)(1 - p) * (Bb * MEM);
#pragma unroll
    for (int e = 0; e < 4; ++e) {
      const int idx = e * 128 + tid;          // 0..511
      const int b   = idx >> 3;               // batch row 0..63
      const int jl  = idx & 7;                // local col 0..7
      const int j   = wg * 8 + jl;            // h column
      const float ma = stage[b * 17 + jl];
      const float mc = stage[b * 17 + jl + 8];
      const size_t eb = (size_t)(b * Ll + t) * (3 * MEM);
      const float ia = (float)emb[eb + j];
      const float ic = (float)emb[eb + MEM + j];
      const float io = (float)emb[eb + 2 * MEM + j];
      const float hv = hreg[e];
      const float av = 1.0f + tanhf(ia + ma);
      const float cv = 1.0f / (1.0f + __expf(-(ic + mc)));
      const float hn = cv * hv + (1.0f - cv) * tanhf(io + av * hv);
      hreg[e] = hn;
      hwrite[(size_t)b * MEM + j] = (_Float16)hn;
      out[(size_t)(b * Ll + t) * MEM + j] = hn;            // h_seq
      if (t == Ll - 1) out[SEQ_ELEMS + (size_t)b * MEM + j] = hn;  // h_last
    }

    grid_barrier(bar);                        // publish hwrite, protect stage/ldsB reuse
    p ^= 1;
  }
}

// ---------------------------------------------------------------------------
extern "C" void kernel_launch(void* const* d_in, const int* in_sizes, int n_in,
                              void* d_out, int out_size, void* d_ws, size_t ws_size,
                              hipStream_t stream) {
  (void)in_sizes; (void)n_in; (void)out_size; (void)ws_size;  // ws needs ~243 MB
  const float* u    = (const float*)d_in[0];
  const float* W_im = (const float*)d_in[1];
  const float* W_mm = (const float*)d_in[2];
  float* out = (float*)d_out;

  char* ws = (char*)d_ws;
  size_t off = 0;
  auto carve = [&](size_t bytes) -> char* {
    char* p = ws + off;
    off += (bytes + 255) & ~(size_t)255;
    return p;
  };
  _Float16* u16   = (_Float16*)carve(U_ELEMS   * sizeof(_Float16));
  _Float16* wim16 = (_Float16*)carve(WIM_ELEMS * sizeof(_Float16));
  _Float16* wmm16 = (_Float16*)carve(WMM_ELEMS * sizeof(_Float16));
  _Float16* emb16 = (_Float16*)carve(EMB_ELEMS * sizeof(_Float16));
  _Float16* hbuf  = (_Float16*)carve(H_ELEMS   * sizeof(_Float16));
  unsigned* bar   = (unsigned*)carve(2 * sizeof(unsigned));

  k_cvt<<<8192, 256, 0, stream>>>(u,    u16,   U_ELEMS);
  k_cvt<<<2048, 256, 0, stream>>>(W_im, wim16, WIM_ELEMS);
  k_cvt<<<2048, 256, 0, stream>>>(W_mm, wmm16, WMM_ELEMS);
  k_init<<<(unsigned)((H_ELEMS + 255) / 256), 256, 0, stream>>>(hbuf, bar);

  k_proj<<<dim3(512, 96), 256, 0, stream>>>(u16, wim16, emb16);   // 32768 x 3072 x 512
  k_rec<<<NWG, 128, 0, stream>>>(wmm16, emb16, hbuf, out, bar);   // 512 sequential steps
}
